// GroupedQueryAttention_21835613733519
// MI455X (gfx1250) — matrix operand order repaired and checked by reading, not compiled, not run on hardware
//
#include <hip/hip_runtime.h>
#include <hip/hip_bf16.h>
#include <math.h>

// ---------------------------------------------------------------------------
// GQA forward for MI455X (gfx1250): all GEMMs + attention via
// v_wmma_f32_16x16x32_f16 (wave32). Flash attention with 64-key tiles,
// double-buffered GLOBAL_LOAD_ASYNC_TO_LDS_B128 K/V staging shared by the
// 4 waves of a block, ones-WMMA row sums, ds_swizzle max reductions.
// ---------------------------------------------------------------------------

typedef __attribute__((ext_vector_type(16))) _Float16 v16h;
typedef __attribute__((ext_vector_type(8)))  _Float16 v8h;
typedef __attribute__((ext_vector_type(4)))  _Float16 v4h;
typedef __attribute__((ext_vector_type(8)))  float    v8f;
typedef __attribute__((ext_vector_type(4)))  float    v4f;

#define S_LEN   4096
#define D_MODEL 1024
#define N_HEADS 16
#define D_K     64
#define KV_DIM  256   // 4 kv heads * 64
#define KT      64    // attention key-tile size
#define N_KT    (S_LEN / KT)

__device__ __forceinline__ v16h mk16(v8h a, v8h b) {
    v16h r;
#pragma unroll
    for (int i = 0; i < 8; ++i) { r[i] = a[i]; r[i + 8] = b[i]; }
    return r;
}

__device__ __forceinline__ v8f wmma_f16(v16h a, v16h b, v8f c) {
    return __builtin_amdgcn_wmma_f32_16x16x32_f16(false, a, false, b, (short)0, c,
                                                  false, false);
}

// Issue one per-lane async b128 copy global -> LDS (tracked by ASYNCcnt).
__device__ __forceinline__ void async_b128(const void* gaddr, const void* laddr) {
    const unsigned long long ga = (unsigned long long)gaddr;
    const unsigned           la = (unsigned)(unsigned long long)laddr;
    asm volatile("global_load_async_to_lds_b128 %0, %1, off"
                 :: "v"(la), "v"(ga) : "memory");
}

// Max-reduction across the 16-lane half-groups via ds_swizzle XOR patterns
// (and_mask=0x1f, xor_mask=m -> offset = m<<10 | 0x1f).
__device__ __forceinline__ float xmax16(float x) {
    x = fmaxf(x, __int_as_float(__builtin_amdgcn_ds_swizzle(__float_as_int(x), 0x041f)));
    x = fmaxf(x, __int_as_float(__builtin_amdgcn_ds_swizzle(__float_as_int(x), 0x081f)));
    x = fmaxf(x, __int_as_float(__builtin_amdgcn_ds_swizzle(__float_as_int(x), 0x101f)));
    x = fmaxf(x, __int_as_float(__builtin_amdgcn_ds_swizzle(__float_as_int(x), 0x201f)));
    return x;
}

// ---------------------------------------------------------------------------
// Generic GEMM:  C[M x N] = f16(A[M x K]) @ f16(W[K x N]) + bias
//   A, W, bias f32 in global; converted to f16 on the LDS-staging path.
//   Block = 256 threads (8 waves), tile 128(M) x 64(N), K-step 32.
// MODE 0: out f16 row-major | MODE 1: out f16 transposed | MODE 2: out f32
// ---------------------------------------------------------------------------
template <int MODE>
__global__ __launch_bounds__(256)
void gemm_wmma_f16(const float* __restrict__ A, const float* __restrict__ W,
                   const float* __restrict__ bias, void* __restrict__ Cout,
                   int M, int N, int K, int ldc) {
    __shared__ _Float16 la[128 * 32];   // A tile, row-major [128][32]
    __shared__ _Float16 lw[64 * 32];    // W tile, TRANSPOSED [n][k]

    const int tid  = threadIdx.x;
    const int lane = tid & 31;
    const int wv   = tid >> 5;
    const int l15  = lane & 15;
    const int hi   = lane >> 4;
    const int mBase = blockIdx.y * 128;
    const int nBase = blockIdx.x * 64;

    v8f acc[4];
#pragma unroll
    for (int c = 0; c < 4; ++c) acc[c] = (v8f)0.0f;

    for (int k0 = 0; k0 < K; k0 += 32) {
        __syncthreads();
#pragma unroll
        for (int p = 0; p < 4; ++p) {
            const int row = (tid >> 3) + p * 32;
            const int kc  = (tid & 7) * 4;
            v4f t = *(const v4f*)(A + (size_t)(mBase + row) * K + k0 + kc);
            v4h s;
#pragma unroll
            for (int j = 0; j < 4; ++j) s[j] = (_Float16)t[j];
            *(v4h*)(la + row * 32 + kc) = s;
        }
#pragma unroll
        for (int p = 0; p < 2; ++p) {
            const int kr = (tid >> 4) + p * 16;
            const int nc = (tid & 15) * 4;
            v4f t = *(const v4f*)(W + (size_t)(k0 + kr) * N + nBase + nc);
#pragma unroll
            for (int j = 0; j < 4; ++j)
                lw[(nc + j) * 32 + kr] = (_Float16)t[j];
        }
        __syncthreads();

        const int arow = wv * 16 + l15;
        const int ka   = hi ? 16 : 0;
        v16h aF = mk16(*(const v8h*)(la + arow * 32 + ka),
                       *(const v8h*)(la + arow * 32 + ka + 8));
#pragma unroll
        for (int cc = 0; cc < 4; ++cc) {
            const int col = cc * 16 + l15;
            const int kb  = hi ? 16 : 0;
            v16h bF = *(const v16h*)(lw + col * 32 + kb);
            acc[cc] = wmma_f16(aF, bF, acc[cc]);
        }
    }

    const int rowBase = mBase + wv * 16 + hi * 8;
#pragma unroll
    for (int cc = 0; cc < 4; ++cc) {
        const int col = nBase + cc * 16 + l15;
        const float bv = bias[col];
        if (MODE == 1) {
            v8h pk;
#pragma unroll
            for (int v = 0; v < 8; ++v) pk[v] = (_Float16)(acc[cc][v] + bv);
            *(v8h*)((_Float16*)Cout + (size_t)col * ldc + rowBase) = pk;
        } else {
#pragma unroll
            for (int v = 0; v < 8; ++v) {
                const float r = acc[cc][v] + bv;
                const size_t idx = (size_t)(rowBase + v) * ldc + col;
                if (MODE == 0) ((_Float16*)Cout)[idx] = (_Float16)r;
                else           ((float*)Cout)[idx]    = r;
            }
        }
    }
}

// ---------------------------------------------------------------------------
// Flash GQA attention. Block = 4 waves, all same head; wave owns 16 q rows.
// 64-key tiles async-copied to LDS (double buffered, shared by the block):
// K as [64 key][64 d], V as [64 d][64 key]. Per tile per wave: 8 QK^T WMMAs,
// one 4-stage ds_swizzle max reduction, exp2, P via private LDS tile,
// 8 PV WMMAs + 2 ones-WMMAs that carry the softmax denominator.
// ---------------------------------------------------------------------------
__global__ __launch_bounds__(128)
void gqa_flash_wmma(const _Float16* __restrict__ q,
                    const _Float16* __restrict__ kf,
                    const _Float16* __restrict__ vt,
                    float* __restrict__ attn) {
    __shared__ _Float16 Kt_[2][KT * 64];   // [buf][key][d]   8KB each
    __shared__ _Float16 Vt_[2][64 * KT];   // [buf][d][key]   8KB each
    __shared__ _Float16 Pb[4][16 * KT];    // per-wave P tile (16 q x 64 keys)

    const int tid  = threadIdx.x;
    const int lane = tid & 31;
    const int wv   = tid >> 5;
    const int l15  = lane & 15;
    const int hi   = lane >> 4;
    const int h    = blockIdx.y;
    const int qt   = blockIdx.x * 4 + wv;
    const int qbase = qt * 16;
    const int g    = h >> 2;               // kv head = h / N_REP
    _Float16* P = Pb[wv];

    // ---- Q A-fragments (row = l15, chunks at hi*16 and +8), d = 0..63 ----
    const _Float16* qrow = q + (size_t)(qbase + l15) * D_MODEL + h * D_K;
    const int ka = hi * 16;
    v16h qA[2];
#pragma unroll
    for (int st = 0; st < 2; ++st)
        qA[st] = mk16(*(const v8h*)(qrow + st * 32 + ka),
                      *(const v8h*)(qrow + st * 32 + ka + 8));

    v16h onesF;
#pragma unroll
    for (int i = 0; i < 16; ++i) onesF[i] = (_Float16)1.0f;

    v8f o0 = (v8f)0.0f, o1 = (v8f)0.0f, o2 = (v8f)0.0f, o3 = (v8f)0.0f;
    v8f oL = (v8f)0.0f;                    // P @ ones: running softmax denom
    float ms[8];
#pragma unroll
    for (int v = 0; v < 8; ++v) ms[v] = -1.0e30f;
    const float C2 = 0.125f * 1.44269504088896f;   // scale * log2(e)

    // ---- issue async K/V tile -> LDS buffer (8 b128 instrs per wave) ----
    auto issue_tile = [&](int kb, int buf) {
#pragma unroll
        for (int j = 0; j < 4; ++j) {   // K: e in [0,512): key = e>>3, d8 = e&7
            const int e = tid + j * 128;
            async_b128(kf + (size_t)(kb + (e >> 3)) * KV_DIM + g * D_K + (e & 7) * 8,
                       Kt_[buf] + e * 8);
        }
#pragma unroll
        for (int j = 0; j < 4; ++j) {   // V: e in [0,512): d = e>>3, k8 = e&7
            const int e = tid + j * 128;
            async_b128(vt + (size_t)(g * D_K + (e >> 3)) * S_LEN + kb + (e & 7) * 8,
                       Vt_[buf] + e * 8);
        }
    };

    issue_tile(0, 0);

    for (int it = 0; it < N_KT; ++it) {
        const int kb = it * KT;
        __syncthreads();   // all waves done reading buf[(it+1)&1] last round
        if (it + 1 < N_KT) {
            issue_tile(kb + KT, (it + 1) & 1);
            asm volatile("s_wait_asynccnt 0x8" ::: "memory");  // tile it done
        } else {
            asm volatile("s_wait_asynccnt 0x0" ::: "memory");
        }
        __syncthreads();   // everyone's tile-it async data visible in LDS
        if (kb + 2 * KT < S_LEN)   // warm L2 for tile it+2
            __builtin_prefetch(kf + (size_t)(kb + 2 * KT + lane) * KV_DIM + g * D_K, 0, 2);

        const _Float16* lk = Kt_[it & 1];
        const _Float16* lv = Vt_[it & 1];

        // ---------- scores: four 16-key column tiles, k = 64 ----------
        v8f s0 = (v8f)0.0f, s1 = (v8f)0.0f, s2 = (v8f)0.0f, s3 = (v8f)0.0f;
#pragma unroll
        for (int st = 0; st < 2; ++st) {
            const int d0 = st * 32 + hi * 16;
            s0 = wmma_f16(qA[st], *(const v16h*)(lk + (0 * 16 + l15) * 64 + d0), s0);
            s1 = wmma_f16(qA[st], *(const v16h*)(lk + (1 * 16 + l15) * 64 + d0), s1);
            s2 = wmma_f16(qA[st], *(const v16h*)(lk + (2 * 16 + l15) * 64 + d0), s2);
            s3 = wmma_f16(qA[st], *(const v16h*)(lk + (3 * 16 + l15) * 64 + d0), s3);
        }

        // ---------- online softmax: one max reduction per 64 keys ----------
#pragma unroll
        for (int v = 0; v < 8; ++v) {
            const float tm = xmax16(fmaxf(fmaxf(s0[v], s1[v]), fmaxf(s2[v], s3[v])));
            const float mn = fmaxf(ms[v], tm);
            const float al = exp2f((ms[v] - mn) * C2);
            ms[v] = mn;
            const float e0 = exp2f((s0[v] - mn) * C2);
            const float e1 = exp2f((s1[v] - mn) * C2);
            const float e2 = exp2f((s2[v] - mn) * C2);
            const float e3 = exp2f((s3[v] - mn) * C2);
            o0[v] *= al; o1[v] *= al; o2[v] *= al; o3[v] *= al; oL[v] *= al;
            const int r = v + hi * 8;      // re-layout P: C-frag -> LDS tile
            P[r * KT +  0 + l15] = (_Float16)e0;
            P[r * KT + 16 + l15] = (_Float16)e1;
            P[r * KT + 32 + l15] = (_Float16)e2;
            P[r * KT + 48 + l15] = (_Float16)e3;
        }
        asm volatile("s_wait_dscnt 0" ::: "memory");  // same-wave LDS RAW

        // ---------- P A-fragments (16 x 64 -> two k=32 fragments) ----------
        const _Float16* pr = P + l15 * KT + hi * 16;
        v16h pA0 = mk16(*(const v8h*)(pr),      *(const v8h*)(pr + 8));
        v16h pA1 = mk16(*(const v8h*)(pr + 32), *(const v8h*)(pr + 40));

        // ---------- O += P @ V, denom += P @ ones ----------
#pragma unroll
        for (int st = 0; st < 2; ++st) {
            const v16h pA = st ? pA1 : pA0;
            const int k0 = st * 32 + hi * 16;
            o0 = wmma_f16(pA, *(const v16h*)(lv + (0 * 16 + l15) * KT + k0), o0);
            o1 = wmma_f16(pA, *(const v16h*)(lv + (1 * 16 + l15) * KT + k0), o1);
            o2 = wmma_f16(pA, *(const v16h*)(lv + (2 * 16 + l15) * KT + k0), o2);
            o3 = wmma_f16(pA, *(const v16h*)(lv + (3 * 16 + l15) * KT + k0), o3);
            oL = wmma_f16(pA, onesF, oL);
        }
    }

    // ---------- normalize and write f32 attention output ----------
#pragma unroll
    for (int v = 0; v < 8; ++v) {
        const float inv = 1.0f / oL[v];
        const size_t row = (size_t)(qbase + v + hi * 8);
        float* orow = attn + row * D_MODEL + h * D_K;
        orow[0 * 16 + l15] = o0[v] * inv;
        orow[1 * 16 + l15] = o1[v] * inv;
        orow[2 * 16 + l15] = o2[v] * inv;
        orow[3 * 16 + l15] = o3[v] * inv;
    }
}

// ---------------------------------------------------------------------------
// Launch. Workspace: qh(8MB) | kh(2MB) | vt(2MB) | attn f32(16MB) = 28MB.
// ---------------------------------------------------------------------------
extern "C" void kernel_launch(void* const* d_in, const int* in_sizes, int n_in,
                              void* d_out, int out_size, void* d_ws, size_t ws_size,
                              hipStream_t stream) {
    const float* query = (const float*)d_in[0];
    const float* key   = (const float*)d_in[1];
    const float* value = (const float*)d_in[2];
    const float* w_q   = (const float*)d_in[3];
    const float* b_q   = (const float*)d_in[4];
    const float* w_k   = (const float*)d_in[5];
    const float* b_k   = (const float*)d_in[6];
    const float* w_v   = (const float*)d_in[7];
    const float* b_v   = (const float*)d_in[8];
    const float* w_o   = (const float*)d_in[9];
    const float* b_o   = (const float*)d_in[10];
    float* out = (float*)d_out;

    _Float16* qh = (_Float16*)d_ws;                        // [4096][1024]
    _Float16* kh = qh + (size_t)S_LEN * D_MODEL;           // [4096][256]
    _Float16* vt = kh + (size_t)S_LEN * KV_DIM;            // [256][4096]
    float*  attn = (float*)(vt + (size_t)S_LEN * KV_DIM);  // [4096][1024]

    const dim3 blk(256);
    gemm_wmma_f16<0><<<dim3(D_MODEL / 64, S_LEN / 128), blk, 0, stream>>>(
        query, w_q, b_q, qh, S_LEN, D_MODEL, D_MODEL, D_MODEL);
    gemm_wmma_f16<0><<<dim3(KV_DIM / 64, S_LEN / 128), blk, 0, stream>>>(
        key, w_k, b_k, kh, S_LEN, KV_DIM, D_MODEL, KV_DIM);
    gemm_wmma_f16<1><<<dim3(KV_DIM / 64, S_LEN / 128), blk, 0, stream>>>(
        value, w_v, b_v, vt, S_LEN, KV_DIM, D_MODEL, S_LEN);
    gqa_flash_wmma<<<dim3(S_LEN / 16 / 4, N_HEADS), dim3(128), 0, stream>>>(
        qh, kh, vt, attn);
    gemm_wmma_f16<2><<<dim3(D_MODEL / 64, S_LEN / 128), blk, 0, stream>>>(
        attn, w_o, b_o, out, S_LEN, D_MODEL, D_MODEL, D_MODEL);
}
